// Gate_65481071394963
// MI455X (gfx1250) — compile-verified, matrix operand order checked by reading, not run on hardware
//
#include <hip/hip_runtime.h>
#include <hip/hip_bf16.h>

#define TOKENS 8192
#define DIM 4096
#define NEXP 256
#define NGRP 8
#define GSIZE 32
#define TOPKG 4
#define TOPK 8
#define ROUTE_SCALE 2.5f
#define MTOK 64   // tokens per workgroup
#define SSTR 260  // padded LDS row stride (floats): 8*260 % 64 == 32 -> lane halves hit different banks

typedef __bf16 v16bf __attribute__((ext_vector_type(16)));
typedef __bf16 v8bf  __attribute__((ext_vector_type(8)));
typedef __bf16 v4bf  __attribute__((ext_vector_type(4)));
typedef float  v8f   __attribute__((ext_vector_type(8)));

__device__ __forceinline__ float sigmoidf(float v) {
    return 1.0f / (1.0f + __expf(-v));
}

// A-fragment (16x32 bf16) per-lane layout for v_wmma_f32_16x16x32_bf16:
// lane r (both halves) holds row M=r; half h holds K = {8h..8h+7} U {16+8h..16+8h+7}.
// B (32x16) layout is identical with N=r when B is supplied as W[n][k] row-major.
__device__ __forceinline__ v16bf load_frag_f32(const float* __restrict__ row, int kc) {
    const float4 f0 = *(const float4*)(row + kc);
    const float4 f1 = *(const float4*)(row + kc + 4);
    const float4 f2 = *(const float4*)(row + kc + 16);
    const float4 f3 = *(const float4*)(row + kc + 20);
    v16bf a;
    a[0]  = (__bf16)f0.x; a[1]  = (__bf16)f0.y; a[2]  = (__bf16)f0.z; a[3]  = (__bf16)f0.w;
    a[4]  = (__bf16)f1.x; a[5]  = (__bf16)f1.y; a[6]  = (__bf16)f1.z; a[7]  = (__bf16)f1.w;
    a[8]  = (__bf16)f2.x; a[9]  = (__bf16)f2.y; a[10] = (__bf16)f2.z; a[11] = (__bf16)f2.w;
    a[12] = (__bf16)f3.x; a[13] = (__bf16)f3.y; a[14] = (__bf16)f3.z; a[15] = (__bf16)f3.w;
    return a;
}

__device__ __forceinline__ v16bf load_frag_bf16(const __bf16* __restrict__ row, int kc) {
    v8bf lo = *(const v8bf*)(row + kc);
    v8bf hi = *(const v8bf*)(row + kc + 16);
    return __builtin_shufflevector(lo, hi, 0, 1, 2, 3, 4, 5, 6, 7,
                                   8, 9, 10, 11, 12, 13, 14, 15);
}

// ---- W fp32 -> bf16 pre-conversion (2 MB result; lives in L2 across the GEMM) ----
__global__ __launch_bounds__(256) void cvt_w_kernel(const float* __restrict__ W,
                                                    void* __restrict__ Wb_) {
    __bf16* __restrict__ Wb = (__bf16*)Wb_;
    int i = (blockIdx.x * 256 + threadIdx.x) * 4;
    float4 f = *(const float4*)(W + i);
    v4bf o = {(__bf16)f.x, (__bf16)f.y, (__bf16)f.z, (__bf16)f.w};
    *(v4bf*)(Wb + i) = o;
}

// ---- Fused scoring GEMM (bf16 WMMA) + sigmoid + group-limited top-k routing ----
// WG = 64 tokens x 256 experts. Wave tile = 32 tokens x 64 experts
// (2 M-frags x 4 N-frags = 8 WMMA per K-step; A-cvt amortized over 4 N tiles).
__global__ __launch_bounds__(256) void gate_kernel(const float* __restrict__ x,
                                                   const void* __restrict__ Wb_,
                                                   const float* __restrict__ bias,
                                                   float* __restrict__ out) {
    const __bf16* __restrict__ Wb = (const __bf16*)Wb_;
    __shared__ float s_sc[MTOK * SSTR];   // biased scores: sigmoid(logit) + bias[e]
    __shared__ float s_bias[NEXP];

    const int tid  = threadIdx.x;
    const int wave = tid >> 5;
    const int lane = tid & 31;
    const int half = lane >> 4;
    const int r    = lane & 15;
    const int mt   = wave >> 2;         // 0..1 : which 32-token half
    const int nb   = (wave & 3) * 64;   // 0,64,128,192 : expert base
    const int m0   = blockIdx.x * MTOK;

    s_bias[tid] = bias[tid];

    const float* xrow0 = x + (size_t)(m0 + mt * 32 + r) * DIM;
    const float* xrow1 = x + (size_t)(m0 + mt * 32 + 16 + r) * DIM;

    const __bf16* wrow[4];
    float bn[4];
#pragma unroll
    for (int n = 0; n < 4; ++n) {
        wrow[n] = Wb + (size_t)(nb + n * 16 + r) * DIM;
        bn[n]   = bias[nb + n * 16 + r];
    }

    v8f acc[2][4];
#pragma unroll
    for (int ml = 0; ml < 2; ++ml)
#pragma unroll
        for (int n = 0; n < 4; ++n) acc[ml][n] = (v8f){};

    const int kh = half * 8;

#pragma unroll 2
    for (int k0 = 0; k0 < DIM; k0 += 32) {
        const int kc = k0 + kh;
        v16bf a0 = load_frag_f32(xrow0, kc);
        v16bf a1 = load_frag_f32(xrow1, kc);
#pragma unroll
        for (int n = 0; n < 4; ++n) {
            v16bf bf = load_frag_bf16(wrow[n], kc);
            acc[0][n] = __builtin_amdgcn_wmma_f32_16x16x32_bf16(false, a0, false, bf, (short)0, acc[0][n], false, false);
            acc[1][n] = __builtin_amdgcn_wmma_f32_16x16x32_bf16(false, a1, false, bf, (short)0, acc[1][n], false, false);
        }
    }

    // C layout: VGPR j -> row j (lanes 0-15) / row j+8 (lanes 16-31), col = lane&15.
#pragma unroll
    for (int ml = 0; ml < 2; ++ml) {
#pragma unroll
        for (int n = 0; n < 4; ++n) {
            const int col = nb + n * 16 + r;
#pragma unroll
            for (int j = 0; j < 8; ++j) {
                const int row = mt * 32 + ml * 16 + half * 8 + j;
                s_sc[row * SSTR + col] = sigmoidf(acc[ml][n][j]) + bn[n];
            }
        }
    }
    __syncthreads();

    // ---- routing: one thread per token (64 tokens / block -> waves 0-1;
    //      other waves fall through past the EXECZ branch and exit) ----
    if (tid < MTOK) {
        const float* srow = &s_sc[tid * SSTR];

        // per-group top-2 sum of biased scores
        float gs[NGRP];
#pragma unroll
        for (int g = 0; g < NGRP; ++g) {
            float m1 = -1e30f, m2 = -1e30f;
#pragma unroll 1
            for (int j = 0; j < GSIZE; ++j) {
                float v = srow[g * GSIZE + j];
                if (v > m1) { m2 = m1; m1 = v; }
                else if (v > m2) { m2 = v; }
            }
            gs[g] = m1 + m2;
        }

        // top-4 groups (strict > keeps first-index tie behavior)
        unsigned sel = 0;
#pragma unroll
        for (int p = 0; p < TOPKG; ++p) {
            float best = -1e30f; int bi = 0;
#pragma unroll
            for (int g = 0; g < NGRP; ++g) {
                bool avail = ((sel >> g) & 1u) == 0u;
                if (avail && gs[g] > best) { best = gs[g]; bi = g; }
            }
            sel |= (1u << bi);
        }

        // top-8 over the 128 candidates in the selected groups (stable insertion)
        float tv[TOPK]; int ti[TOPK];
#pragma unroll
        for (int q = 0; q < TOPK; ++q) { tv[q] = -1e30f; ti[q] = 0; }
#pragma unroll
        for (int g = 0; g < NGRP; ++g) {
            if ((sel >> g) & 1u) {
#pragma unroll 1
                for (int j = 0; j < GSIZE; ++j) {
                    const int e = g * GSIZE + j;
                    float cv = srow[e];
                    int   ci = e;
#pragma unroll
                    for (int q = 0; q < TOPK; ++q) {
                        if (cv > tv[q]) {
                            float fv = tv[q]; int fi = ti[q];
                            tv[q] = cv; ti[q] = ci;
                            cv = fv; ci = fi;
                        }
                    }
                }
            }
        }

        // weights = original sigmoid scores (biased - bias), renormalized * scale
        float wv[TOPK];
        float wsum = 0.0f;
#pragma unroll
        for (int q = 0; q < TOPK; ++q) {
            wv[q] = srow[ti[q]] - s_bias[ti[q]];
            wsum += wv[q];
        }
        const float inv = ROUTE_SCALE / wsum;

        const int token = m0 + tid;
        float* outw = out + (size_t)token * TOPK;
        int*   outi = (int*)out + (size_t)TOKENS * TOPK + (size_t)token * TOPK;
#pragma unroll
        for (int q = 0; q < TOPK; ++q) {
            outw[q] = wv[q] * inv;
            outi[q] = ti[q];
        }
    }
}

extern "C" void kernel_launch(void* const* d_in, const int* in_sizes, int n_in,
                              void* d_out, int out_size, void* d_ws, size_t ws_size,
                              hipStream_t stream) {
    (void)in_sizes; (void)n_in; (void)out_size; (void)ws_size;
    const float* x    = (const float*)d_in[0];
    const float* W    = (const float*)d_in[1];
    const float* bias = (const float*)d_in[2];

    // ws: bf16 copy of W (256*4096*2 = 2 MB)
    void* Wb = d_ws;

    cvt_w_kernel<<<(NEXP * DIM) / (4 * 256), 256, 0, stream>>>(W, Wb);
    gate_kernel<<<TOKENS / MTOK, 256, 0, stream>>>(x, Wb, bias, (float*)d_out);
}
